// MODEL_14920716387043
// MI455X (gfx1250) — matrix operand order repaired and checked by reading, not compiled, not run on hardware
//
#include <hip/hip_runtime.h>

// ---------------- problem constants (match reference) ----------------
#define BB     4
#define TLEN   4096
#define NTOK   (BB * TLEN)      // 16384
#define DMODEL 512
#define NH     8
#define DQ     64
#define DV     64
#define DFF    1024
#define NWAVLM 13

#define ACT_NONE 0
#define ACT_RELU 1
#define ACT_ELU1 2

typedef __attribute__((ext_vector_type(16))) __bf16       bf16x16;
typedef __attribute__((ext_vector_type(8)))  float        f32x8;
typedef __attribute__((ext_vector_type(4)))  unsigned int u32x4v;
typedef __attribute__((ext_vector_type(8)))  int          i32x8v;
typedef __attribute__((ext_vector_type(4)))  int          i32x4v;

__device__ __forceinline__ unsigned short f32_to_bf16(float f) {
    unsigned int u = __float_as_uint(f);
    u += 0x7FFFu + ((u >> 16) & 1u);          // round-to-nearest-even
    return (unsigned short)(u >> 16);
}

// ---------------------------------------------------------------------
// TDM 2D tile load: bf16 tensor [dim1, dim0] row-major (stride0 elems),
// tile = 128 rows x 32 cols starting at gaddr, into LDS at lds_off with
// 4-DWORD padding after every 16 DWORDs (64B row) -> LDS row stride 40 bf16.
// Descriptor per CDNA5 ISA 8.3/8.4 (group0 128b, group1 256b), groups 2/3 zero.
// ---------------------------------------------------------------------
__device__ __forceinline__ void tdm_load_tile(unsigned long long gaddr,
                                              unsigned lds_off,
                                              unsigned dim0, unsigned dim1,
                                              unsigned stride0) {
    u32x4v g0 = {};
    g0[0] = 1u;                                        // count=1 (valid), user desc
    g0[1] = lds_off;                                   // lds_addr (bytes)
    g0[2] = (unsigned)gaddr;                           // global_addr[31:0]
    g0[3] = (unsigned)((gaddr >> 32) & 0x1FFFFFFull)   // global_addr[56:32]
          | (2u << 30);                                // type=2 ("image")
    i32x8v g1 = {};
    g1[0] = (int)((1u << 16)        // data_size = 2 bytes
                | (1u << 20)        // pad_enable
                | (3u << 22)        // pad_interval: 16 DWORDs (one 64B row)
                | (3u << 25));      // pad_amount:   4 DWORDs (8 bf16)
    g1[1] = (int)((dim0 & 0xFFFFu) << 16);                               // tensor_dim0[15:0]
    g1[2] = (int)(((dim0 >> 16) & 0xFFFFu) | ((dim1 & 0xFFFFu) << 16));  // dim0[31:16] | dim1[15:0]
    g1[3] = (int)(((dim1 >> 16) & 0xFFFFu) | (32u << 16));               // dim1[31:16] | tile_dim0=32
    g1[4] = (int)128u;                                                   // tile_dim1=128, tile_dim2=0
    g1[5] = (int)stride0;                                                // tensor_dim0_stride[31:0]
    g1[6] = 0;                                                           // stride hi | dim1_stride lo
    g1[7] = 0;
    i32x4v z4 = {};
#if defined(__clang_major__) && (__clang_major__ >= 23)
    i32x8v z8 = {};
    __builtin_amdgcn_tensor_load_to_lds(g0, g1, z4, z4, z8, 0);
#else
    __builtin_amdgcn_tensor_load_to_lds(g0, g1, z4, z4, 0);
#endif
}

// ---------------------------------------------------------------------
// Tiled WMMA GEMM:  C[M,N] = act(A[M,K] @ B[K,N] + bias)
// A: bf16 [M,K] row-major. Bt: bf16 [N,K] row-major (pre-transposed B).
// TDM double-buffered LDS staging, f32 accum via v_wmma_f32_16x16x32_bf16.
// Block: 256 threads (8 wave32), C tile 128x128, K step 32.
// Requires M%128==0, N%128==0, K%32==0 (true for all GEMMs here).
// Outputs: Cf (fp32, optional) and Cb (bf16, optional), activation fused.
// ---------------------------------------------------------------------
__launch_bounds__(256)
__global__ void wmma_gemm_tdm_kernel(const unsigned short* __restrict__ A,
                                     const unsigned short* __restrict__ Bt,
                                     const float* __restrict__ bias,
                                     float* __restrict__ Cf,
                                     unsigned short* __restrict__ Cb,
                                     int M, int N, int Kdim, int act)
{
    __shared__ __align__(16) unsigned short sa[2][128 * 40];
    __shared__ __align__(16) unsigned short sb[2][128 * 40];

    const int tid  = threadIdx.x;
    const int lane = tid & 31;
    const int wid  = tid >> 5;
    const int wm   = wid & 1;    // 2 wave rows  (64 M each)
    const int wn   = wid >> 1;   // 4 wave cols  (32 N each)
    const int m0   = blockIdx.y * 128;
    const int n0   = blockIdx.x * 128;

    const unsigned lds_sa[2] = { (unsigned)(size_t)(void*)&sa[0][0],
                                 (unsigned)(size_t)(void*)&sa[1][0] };
    const unsigned lds_sb[2] = { (unsigned)(size_t)(void*)&sb[0][0],
                                 (unsigned)(size_t)(void*)&sb[1][0] };
    const unsigned long long gA = (unsigned long long)(size_t)A;
    const unsigned long long gB = (unsigned long long)(size_t)Bt;

    f32x8 acc[4][2];
    #pragma unroll
    for (int i = 0; i < 4; ++i)
        #pragma unroll
        for (int j = 0; j < 2; ++j) acc[i][j] = {};

    const int nk = Kdim >> 5;   // K tiles of 32

    if (wid == 0) {
        tdm_load_tile(gA + ((size_t)m0 * Kdim) * 2, lds_sa[0], (unsigned)Kdim, (unsigned)M, (unsigned)Kdim);
        tdm_load_tile(gB + ((size_t)n0 * Kdim) * 2, lds_sb[0], (unsigned)Kdim, (unsigned)N, (unsigned)Kdim);
    }

    for (int ki = 0; ki < nk; ++ki) {
        const int cur = ki & 1;
        if (wid == 0) __builtin_amdgcn_s_wait_tensorcnt(0);  // tile ki landed in LDS
        __syncthreads();                                     // visible to all; prev buf free
        if (wid == 0 && (ki + 1) < nk) {                     // prefetch tile ki+1
            const size_t k1 = (size_t)(ki + 1) << 5;
            tdm_load_tile(gA + ((size_t)m0 * Kdim + k1) * 2, lds_sa[cur ^ 1], (unsigned)Kdim, (unsigned)M, (unsigned)Kdim);
            tdm_load_tile(gB + ((size_t)n0 * Kdim + k1) * 2, lds_sb[cur ^ 1], (unsigned)Kdim, (unsigned)N, (unsigned)Kdim);
        }

        union Frag { bf16x16 v; u32x4v q[2]; };
        Frag fa[4], fb[2];

        // A fragment (16x32, 16-bit): lane L row = L&15;
        // elems 0..7 = K kb..kb+7, elems 8..15 = K kb+16..kb+23, kb = (L>=16)?8:0
        const int kb = (lane >> 4) * 8;
        #pragma unroll
        for (int i = 0; i < 4; ++i) {
            int row = wm * 64 + i * 16 + (lane & 15);
            const unsigned short* p = &sa[cur][row * 40 + kb];
            fa[i].q[0] = *reinterpret_cast<const u32x4v*>(p);
            fa[i].q[1] = *reinterpret_cast<const u32x4v*>(p + 16);
        }
        // B fragment (32x16): lane col = lane&15; elems 0..7 = K kbb..kbb+7, elems 8..15 = K kbb+16..kbb+23, kbb = (L>=16)?8:0
        const int kbb = (lane >> 4) * 8;
        #pragma unroll
        for (int j = 0; j < 2; ++j) {
            int col = wn * 32 + j * 16 + (lane & 15);
            const unsigned short* p = &sb[cur][col * 40 + kbb];
            fb[j].q[0] = *reinterpret_cast<const u32x4v*>(p);
            fb[j].q[1] = *reinterpret_cast<const u32x4v*>(p + 16);
        }

        #pragma unroll
        for (int i = 0; i < 4; ++i)
            #pragma unroll
            for (int j = 0; j < 2; ++j)
                acc[i][j] = __builtin_amdgcn_wmma_f32_16x16x32_bf16(
                    false, fa[i].v, false, fb[j].v, (short)0, acc[i][j], false, false);
    }

    // epilogue: C/D layout — VGPR r: lanes 0-15 M=r, lanes 16-31 M=8+r; N = lane&15
    #pragma unroll
    for (int i = 0; i < 4; ++i) {
        #pragma unroll
        for (int j = 0; j < 2; ++j) {
            int colg = n0 + wn * 32 + j * 16 + (lane & 15);
            float bv = bias ? bias[colg] : 0.f;
            #pragma unroll
            for (int r = 0; r < 8; ++r) {
                int rowg = m0 + wm * 64 + i * 16 + r + ((lane >> 4) * 8);
                float v = acc[i][j][r] + bv;
                if (act == ACT_RELU)       v = v > 0.f ? v : 0.f;
                else if (act == ACT_ELU1)  v = v > 0.f ? v + 1.f : __expf(v); // elu(x)+1
                size_t o = (size_t)rowg * N + colg;
                if (Cf) Cf[o] = v;
                if (Cb) Cb[o] = f32_to_bf16(v);
            }
        }
    }
}

// ---------------- small helper kernels ----------------
__global__ void sigmoid13_kernel(const float* __restrict__ lw, float* __restrict__ sw) {
    int i = threadIdx.x;
    if (i < NWAVLM) sw[i] = 1.f / (1.f + __expf(-lw[i]));
}

// Wt[n,k] = bf16(W[k,n])  (weights are tiny; once per launch)
__global__ void transpose_w_bf16_kernel(const float* __restrict__ W,
                                        unsigned short* __restrict__ Wt,
                                        int Kdim, int N) {
    int idx = blockIdx.x * blockDim.x + threadIdx.x;
    if (idx >= Kdim * N) return;
    int k = idx / N, n = idx - k * N;
    Wt[(size_t)n * Kdim + k] = f32_to_bf16(W[idx]);
}

__global__ void embed_mix_kernel(const float* __restrict__ emb,
                                 const float* __restrict__ sw,
                                 float* __restrict__ Xf,
                                 unsigned short* __restrict__ Xb, int n) {
    int idx = blockIdx.x * blockDim.x + threadIdx.x;
    if (idx >= n) return;
    const float* e = emb + (size_t)idx * NWAVLM;
    float s = 0.f;
    #pragma unroll
    for (int l = 0; l < NWAVLM; ++l) s += e[l] * sw[l];
    Xf[idx] = s;
    Xb[idx] = f32_to_bf16(s);
}

// kvT[bh][d][m] = sum_t k[t,d]*v[t,m];  ksum[bh][d] = sum_t k[t,d]
__global__ void attn_reduce_kernel(const float* __restrict__ Kb,
                                   const float* __restrict__ Vb,
                                   float* __restrict__ KVT,
                                   float* __restrict__ KSUM) {
    int bh = blockIdx.x;            // 0..31
    int d  = blockIdx.y;            // 0..63
    int m  = threadIdx.x;           // 0..63
    int b = bh >> 3, h = bh & 7;
    const float* kcol = Kb + ((size_t)b * TLEN) * DMODEL + h * DQ + d;
    const float* vcol = Vb + ((size_t)b * TLEN) * DMODEL + h * DV + m;
    float acc = 0.f, ks = 0.f;
    for (int t = 0; t < TLEN; ++t) {
        float kd = kcol[(size_t)t * DMODEL];
        acc += kd * vcol[(size_t)t * DMODEL];
        ks  += kd;
    }
    KVT[(size_t)bh * DQ * DV + d * DV + m] = acc;
    if (m == 0) KSUM[(size_t)bh * DQ + d] = ks;
}

// out_bf16[t, h*DV+m] = (q(t) . kvT[:,m]) / (q(t) . ksum + eps)
__global__ void attn_out_kernel(const float* __restrict__ Qb,
                                const float* __restrict__ KVT,
                                const float* __restrict__ KSUM,
                                unsigned short* __restrict__ AOb) {
    int t = blockIdx.x, h = blockIdx.y, b = blockIdx.z;
    int m = threadIdx.x;
    int bh = b * NH + h;
    const float* qr = Qb + ((size_t)(b * TLEN + t)) * DMODEL + h * DQ;
    const float* kv = KVT + (size_t)bh * DQ * DV;
    const float* ks = KSUM + (size_t)bh * DQ;
    float s = 0.f, o = 0.f;
    #pragma unroll 4
    for (int d = 0; d < DQ; ++d) {
        float q = qr[d];
        s += q * ks[d];
        o += q * kv[d * DV + m];
    }
    AOb[((size_t)(b * TLEN + t)) * DMODEL + h * DV + m] = f32_to_bf16(o / (s + 1e-6f));
}

// Out = LayerNorm(X + Yv) * g + b   (Yv may be null; Out may alias X; bf16 copy too)
__global__ void residual_ln_kernel(const float* __restrict__ X,
                                   const float* __restrict__ Yv,
                                   const float* __restrict__ g,
                                   const float* __restrict__ bta,
                                   float* __restrict__ Out,
                                   unsigned short* __restrict__ Ob) {
    __shared__ float red[256];
    int row = blockIdx.x;
    int tid = threadIdx.x;
    const float* xr = X + (size_t)row * DMODEL;
    float x0 = xr[tid], x1 = xr[tid + 256];
    if (Yv) {
        x0 += Yv[(size_t)row * DMODEL + tid];
        x1 += Yv[(size_t)row * DMODEL + tid + 256];
    }
    red[tid] = x0 + x1;
    __syncthreads();
    for (int s = 128; s > 0; s >>= 1) { if (tid < s) red[tid] += red[tid + s]; __syncthreads(); }
    float mu = red[0] * (1.f / DMODEL);
    __syncthreads();
    float d0 = x0 - mu, d1 = x1 - mu;
    red[tid] = d0 * d0 + d1 * d1;
    __syncthreads();
    for (int s = 128; s > 0; s >>= 1) { if (tid < s) red[tid] += red[tid + s]; __syncthreads(); }
    float rstd = rsqrtf(red[0] * (1.f / DMODEL) + 1e-5f);
    float v0 = d0 * rstd * g[tid]       + bta[tid];
    float v1 = d1 * rstd * g[tid + 256] + bta[tid + 256];
    float* outr = Out + (size_t)row * DMODEL;
    outr[tid]       = v0;
    outr[tid + 256] = v1;
    unsigned short* obr = Ob + (size_t)row * DMODEL;
    obr[tid]       = f32_to_bf16(v0);
    obr[tid + 256] = f32_to_bf16(v1);
}

__global__ void head_kernel(const float* __restrict__ X,
                            const float* __restrict__ Wh,
                            const float* __restrict__ bh,
                            float* __restrict__ out) {
    __shared__ float red[64];
    int row = blockIdx.x, tid = threadIdx.x;
    float s = 0.f;
    for (int i = tid; i < DMODEL; i += 64) s += X[(size_t)row * DMODEL + i] * Wh[i];
    red[tid] = s;
    __syncthreads();
    for (int st = 32; st > 0; st >>= 1) { if (tid < st) red[tid] += red[tid + st]; __syncthreads(); }
    if (tid == 0) out[row] = red[0] + bh[0];
}

// ---------------------------------------------------------------------
extern "C" void kernel_launch(void* const* d_in, const int* in_sizes, int n_in,
                              void* d_out, int out_size, void* d_ws, size_t ws_size,
                              hipStream_t stream) {
    (void)in_sizes; (void)n_in; (void)out_size; (void)ws_size;

    const float* emb = (const float*)d_in[0];   // [B,T,D,13]
    const float* lw  = (const float*)d_in[1];   // [13]
    auto P = [&](int i) { return (const float*)d_in[i]; };
    // params flatten order: per layer {Wq,bq,Wk,bk,Wv,bv,Wo,bo,ln1_g,ln1_b,ln2_g,ln2_b,W1,b1,W2,b2}
    // then lnf_g(34), lnf_b(35), Wh(36), bh(37)

    char* wsb = (char*)d_ws;
    float* out = (float*)d_out;

    size_t off = 0;
    auto carve = [&](size_t bytes) { size_t o = off; off += (bytes + 255) & ~(size_t)255; return o; };

    float*          Xf   = (float*)(wsb + carve((size_t)NTOK * DMODEL * 4));
    float*          Qf   = (float*)(wsb + carve((size_t)NTOK * DMODEL * 4));
    float*          Kf   = (float*)(wsb + carve((size_t)NTOK * DMODEL * 4));
    float*          Vf   = (float*)(wsb + carve((size_t)NTOK * DMODEL * 4));
    float*          Yf   = (float*)(wsb + carve((size_t)NTOK * DMODEL * 4));
    unsigned short* Xb   = (unsigned short*)(wsb + carve((size_t)NTOK * DMODEL * 2));
    unsigned short* AOb  = (unsigned short*)(wsb + carve((size_t)NTOK * DMODEL * 2));
    unsigned short* Fb   = (unsigned short*)(wsb + carve((size_t)NTOK * DFF * 2));
    float*          KVT  = (float*)(wsb + carve((size_t)32 * DQ * DV * 4));
    float*          KSUM = (float*)(wsb + carve((size_t)32 * DQ * 4));
    float*          SIGW = (float*)(wsb + carve(64));
    // bf16 transposed weights, per layer: Wq,Wk,Wv,Wo [512x512], W1 [1024x512], W2 [512x1024]
    unsigned short* Wt[2][6];
    for (int l = 0; l < 2; ++l) {
        for (int w = 0; w < 4; ++w) Wt[l][w] = (unsigned short*)(wsb + carve((size_t)DMODEL * DMODEL * 2));
        Wt[l][4] = (unsigned short*)(wsb + carve((size_t)DFF * DMODEL * 2));
        Wt[l][5] = (unsigned short*)(wsb + carve((size_t)DMODEL * DFF * 2));
    }

    // ---- weight conversion/transposition (once per launch) ----
    for (int l = 0; l < 2; ++l) {
        int pb = 2 + l * 16;
        int n512  = DMODEL * DMODEL;
        int n1024 = DMODEL * DFF;
        transpose_w_bf16_kernel<<<(n512  + 255) / 256, 256, 0, stream>>>(P(pb + 0),  Wt[l][0], DMODEL, DMODEL);
        transpose_w_bf16_kernel<<<(n512  + 255) / 256, 256, 0, stream>>>(P(pb + 2),  Wt[l][1], DMODEL, DMODEL);
        transpose_w_bf16_kernel<<<(n512  + 255) / 256, 256, 0, stream>>>(P(pb + 4),  Wt[l][2], DMODEL, DMODEL);
        transpose_w_bf16_kernel<<<(n512  + 255) / 256, 256, 0, stream>>>(P(pb + 6),  Wt[l][3], DMODEL, DMODEL);
        transpose_w_bf16_kernel<<<(n1024 + 255) / 256, 256, 0, stream>>>(P(pb + 12), Wt[l][4], DMODEL, DFF);
        transpose_w_bf16_kernel<<<(n1024 + 255) / 256, 256, 0, stream>>>(P(pb + 14), Wt[l][5], DFF, DMODEL);
    }

    sigmoid13_kernel<<<1, 32, 0, stream>>>(lw, SIGW);
    int nmix = NTOK * DMODEL;
    embed_mix_kernel<<<(nmix + 255) / 256, 256, 0, stream>>>(emb, SIGW, Xf, Xb, nmix);

    dim3 g512(DMODEL / 128, NTOK / 128);   // (4, 128)
    dim3 g1024(DFF / 128, NTOK / 128);     // (8, 128)

    for (int l = 0; l < 2; ++l) {
        int pb = 2 + l * 16;
        // projections (elu(x)+1 fused for q,k) -> fp32 for attention math
        wmma_gemm_tdm_kernel<<<g512, 256, 0, stream>>>(Xb, Wt[l][0], P(pb + 1), Qf, nullptr, NTOK, DMODEL, DMODEL, ACT_ELU1);
        wmma_gemm_tdm_kernel<<<g512, 256, 0, stream>>>(Xb, Wt[l][1], P(pb + 3), Kf, nullptr, NTOK, DMODEL, DMODEL, ACT_ELU1);
        wmma_gemm_tdm_kernel<<<g512, 256, 0, stream>>>(Xb, Wt[l][2], P(pb + 5), Vf, nullptr, NTOK, DMODEL, DMODEL, ACT_NONE);
        // linear attention (small einsums)
        attn_reduce_kernel<<<dim3(BB * NH, DQ), DV, 0, stream>>>(Kf, Vf, KVT, KSUM);
        attn_out_kernel<<<dim3(TLEN, NH, BB), DV, 0, stream>>>(Qf, KVT, KSUM, AOb);
        // output projection + residual + LN1
        wmma_gemm_tdm_kernel<<<g512, 256, 0, stream>>>(AOb, Wt[l][3], P(pb + 7), Yf, nullptr, NTOK, DMODEL, DMODEL, ACT_NONE);
        residual_ln_kernel<<<NTOK, 256, 0, stream>>>(Xf, Yf, P(pb + 8), P(pb + 9), Xf, Xb);
        // FFN (hidden kept bf16-only) + residual + LN2
        wmma_gemm_tdm_kernel<<<g1024, 256, 0, stream>>>(Xb, Wt[l][4], P(pb + 13), nullptr, Fb, NTOK, DFF, DMODEL, ACT_RELU);
        wmma_gemm_tdm_kernel<<<g512, 256, 0, stream>>>(Fb, Wt[l][5], P(pb + 15), Yf, nullptr, NTOK, DMODEL, DFF, ACT_NONE);
        residual_ln_kernel<<<NTOK, 256, 0, stream>>>(Xf, Yf, P(pb + 10), P(pb + 11), Xf, Xb);
    }
    // final LN + head
    residual_ln_kernel<<<NTOK, 256, 0, stream>>>(Xf, nullptr, P(34), P(35), Xf, Xb);
    head_kernel<<<NTOK, 64, 0, stream>>>(Xf, P(36), P(37), out);
}
